// LongTermAttention_66331474919991
// MI455X (gfx1250) — matrix operand order repaired and checked by reading, compiled, not run on hardware
//
#include <hip/hip_runtime.h>

#define L_SEQ 2048
#define NB 512
#define DMODEL 1024
#define NH 16
#define DH 64
#define BATCH 8
#define TWO_L 4096
#define INV_SQRT_2PI 0.3989422804014327f

typedef __attribute__((ext_vector_type(16))) __bf16 v16bf;
typedef __attribute__((ext_vector_type(8)))  float  v8f;
typedef __attribute__((ext_vector_type(4)))  unsigned int v4u;
typedef __attribute__((ext_vector_type(8)))  int v8i_;
typedef __attribute__((ext_vector_type(4)))  int v4i_;

union Frag16 { unsigned int u[8]; v16bf v; };

__device__ __forceinline__ unsigned short f2bf(float x){
  unsigned int u = __float_as_uint(x);
  u += 0x7fffu + ((u >> 16) & 1u);
  return (unsigned short)(u >> 16);
}
// A-matrix 16x32 bf16 fragment K-pair base per VGPR i, lane-half h (ISA 7.12.2)
__device__ __forceinline__ int pbA(int i, int h){
  return (i < 4 ? 2*i : 16 + 2*(i-4)) + 8*h;
}
// B-matrix 32x16 bf16 fragment K-pair base per VGPR i, lane-half h (same K order as A)
__device__ __forceinline__ int pbB(int i, int h){
  return (i < 4 ? 2*i : 16 + 2*(i-4)) + 8*h;
}

// ---- Tensor Data Mover: 2D bf16 tile -> LDS with DWORD padding ----
// tile_d0 elements per row (row stride stride0 elements in memory), tile_d1 rows.
// pad 1 DWORD after every 2^(pic+1) DWORDs (pic = pad_interval code).
#define HAVE_TDM __has_builtin(__builtin_amdgcn_tensor_load_to_lds)
#if HAVE_TDM
__device__ __forceinline__ void tdm_load_2d(unsigned lds_off, const void* gptr,
                                            unsigned tile_d0, unsigned tile_d1,
                                            unsigned long long stride0, unsigned pic)
{
  unsigned long long ga = (unsigned long long)(size_t)gptr;
  v4u g0 = { 1u,                                         // count=1, user desc
             lds_off,                                    // lds_addr
             (unsigned)ga,                               // global_addr[31:0]
             (unsigned)((ga >> 32) & 0x01FFFFFFu) | 0x80000000u }; // addr[56:32] | type=2
  // group1 (256b): data_size=2B, pad_enable=1, pad_interval=pic, pad_amount=1 DWORD
  unsigned w0 = (1u << 16) | (1u << 20) | (pic << 22) | (0u << 25);
  unsigned w1 = (tile_d0 & 0xFFFFu) << 16;               // tensor_dim0[15:0]
  unsigned w2 = ((tile_d0 >> 16) & 0xFFFFu) | ((tile_d1 & 0xFFFFu) << 16); // td0 hi | td1 lo
  unsigned w3 = ((tile_d1 >> 16) & 0xFFFFu) | ((tile_d0 & 0xFFFFu) << 16); // td1 hi | tile_dim0
  unsigned w4 = (tile_d1 & 0xFFFFu);                     // tile_dim1, tile_dim2=0
  unsigned w5 = (unsigned)(stride0 & 0xFFFFFFFFull);     // tensor_dim0_stride lo
  unsigned w6 = (unsigned)((stride0 >> 32) & 0xFFFFull); // stride hi, dim1_stride=0
  v8i_ g1 = { (int)w0, (int)w1, (int)w2, (int)w3, (int)w4, (int)w5, (int)w6, 0 };
  v4i_ g2 = { 0, 0, 0, 0 };
  v4i_ g3 = { 0, 0, 0, 0 };
#if __clang_major__ >= 23
  v8i_ g4 = { 0, 0, 0, 0, 0, 0, 0, 0 };
  __builtin_amdgcn_tensor_load_to_lds(g0, g1, g2, g3, g4, 0);
#else
  __builtin_amdgcn_tensor_load_to_lds(g0, g1, g2, g3, 0);
#endif
}
#endif

// ---------------- setup kernels (fp32 VALU) ----------------

__global__ void k_build_F(float* __restrict__ F){ // [512][4096]
  int idx = blockIdx.x*256 + threadIdx.x;
  if (idx >= NB*TWO_L) return;
  int n = idx >> 12;
  int p = idx & 4095;
  const float shift = 1.0f/4096.0f;
  float pos = -0.5f + shift + (float)p * ((2.0f - 2.0f*shift)/4095.0f);
  float mub = (float)(n >> 1) * (1.0f/255.0f);
  float sg  = (n & 1) ? 0.01f : 0.005f;
  float z = (pos - mub)/sg;
  F[idx] = __expf(-0.5f*z*z) * (INV_SQRT_2PI / sg);
}

__global__ void k_gram(const float* __restrict__ F, float* __restrict__ Aug){
  // Aug [512][1024] = [ F F^T + ridge I | I ]
  int idx = blockIdx.x*256 + threadIdx.x;
  if (idx >= NB*1024) return;
  int i = idx >> 10, j = idx & 1023;
  if (j < NB){
    const float* a = F + (long long)i*TWO_L;
    const float* b = F + (long long)j*TWO_L;
    float s = 0.f;
    for (int p = 0; p < TWO_L; p += 4)
      s += a[p]*b[p] + a[p+1]*b[p+1] + a[p+2]*b[p+2] + a[p+3]*b[p+3];
    if (i == j) s += 0.5f;
    Aug[idx] = s;
  } else {
    Aug[idx] = ((j - NB) == i) ? 1.0f : 0.0f;
  }
}

__global__ void k_invert(float* __restrict__ Aug){ // Gauss-Jordan, [512][1024]
  const int t = threadIdx.x; // 1024 threads
  __shared__ float pivinv;
  __shared__ float fac[NB];
  for (int kc = 0; kc < NB; ++kc){
    if (t == 0) pivinv = 1.0f / Aug[kc*1024 + kc];
    __syncthreads();
    Aug[kc*1024 + t] *= pivinv;
    if (t < NB) fac[t] = Aug[t*1024 + kc];
    __syncthreads();
    float pk = Aug[kc*1024 + t];
    for (int i = 0; i < NB; ++i){
      if (i != kc) Aug[i*1024 + t] -= fac[i]*pk;
    }
    __syncthreads();
  }
}

__global__ void k_gsT(const float* __restrict__ F, const float* __restrict__ Aug,
                      unsigned short* __restrict__ GsT){ // [512 n][2048 l] bf16
  int idx = blockIdx.x*256 + threadIdx.x;
  if (idx >= NB*L_SEQ) return;
  int n = idx >> 11, l = idx & 2047;
  float s = 0.f;
  for (int m = 0; m < NB; ++m)
    s += F[(long long)m*TWO_L + 1024 + l] * Aug[m*1024 + NB + n];
  GsT[idx] = f2bf(s);
}

// ---------------- conversion / layout kernels ----------------

__global__ void k_cvt(const float* __restrict__ in, unsigned short* __restrict__ out, long long n){
  long long i = (long long)blockIdx.x*256 + threadIdx.x;
  if (i < n) out[i] = f2bf(in[i]);
}

__global__ void k_cvt_wT(const float* __restrict__ W, unsigned short* __restrict__ WT, float scale){
  int idx = blockIdx.x*256 + threadIdx.x; // 1024*1024, WT[d][e] = W[e][d]*scale
  int d = idx >> 10, e = idx & 1023;
  WT[idx] = f2bf(W[e*1024 + d] * scale);
}

__global__ void k_qflat(const float* __restrict__ query, unsigned short* __restrict__ qf){
  long long idx = (long long)blockIdx.x*256 + threadIdx.x; // B*L*D
  int e = (int)(idx & 1023);
  long long bl = idx >> 10;
  int l = (int)(bl & 2047);
  int b = (int)(bl >> 11);
  int h = e >> 6, c = e & 63;
  qf[idx] = f2bf(query[(((long long)(b*NH + h))*L_SEQ + l)*DH + c]);
}

__global__ void k_split(const unsigned short* __restrict__ X, unsigned short* __restrict__ Y, int rows){
  // X [B][rows][1024] -> Y [B][H][rows][64]
  long long idx = (long long)blockIdx.x*256 + threadIdx.x;
  long long total = (long long)BATCH*rows*DMODEL;
  if (idx >= total) return;
  int e = (int)(idx & 1023);
  long long bl = idx >> 10;
  int rw = (int)(bl % rows);
  int b  = (int)(bl / rows);
  int h = e >> 6, c = e & 63;
  Y[(((long long)(b*NH + h))*rows + rw)*DH + c] = X[idx];
}

__global__ void k_splitT(const unsigned short* __restrict__ X, unsigned short* __restrict__ Y){
  // X [B][512][1024] -> Y [B][H][64][512]
  long long idx = (long long)blockIdx.x*256 + threadIdx.x;
  if (idx >= (long long)BATCH*NB*DMODEL) return;
  int e = (int)(idx & 1023);
  long long bn = idx >> 10;
  int n = (int)(bn & 511);
  int b = (int)(bn >> 9);
  int h = e >> 6, c = e & 63;
  Y[(((long long)(b*NH + h))*DH + c)*NB + n] = X[idx];
}

// ---------------- generic batched bf16 WMMA GEMM (pipelined) ----------------
// C[M,N] (f32 or bf16) = A[M,K] @ B[K,N], row-major, dims multiples of tile.
// Software pipeline: TDM streams A tile kt+1 and global loads stage B tile kt+1
// in registers while WMMAs consume double-buffered LDS tile kt.
__global__ __launch_bounds__(256) void gemm_bf16(
    const unsigned short* __restrict__ A, const unsigned short* __restrict__ B,
    float* __restrict__ Cf, unsigned short* __restrict__ Cb,
    int M, int N, int K, int lda, int ldb, int ldc,
    long long sA, long long sB, long long sC)
{
  __shared__ unsigned short Asl[2][128*34];   // [m][k] padded (TDM pads 1 DW / 16 DW)
  __shared__ unsigned short Bsl[2][128*34];   // [n][k] padded (transposed stage)
  const int tid = threadIdx.x;
  const int lane = tid & 31, wave = tid >> 5;
  const int half = lane >> 4, l16 = lane & 15;
  const int wm = wave & 3, wn = wave >> 2;
  const int bn = blockIdx.x, bm = blockIdx.y, bz = blockIdx.z;
  A += (long long)bz * sA;
  B += (long long)bz * sB;
  (void)M; (void)N;

  const int nk = K >> 5;
  unsigned breg[8];
#if !HAVE_TDM
  unsigned areg[8];
#endif

  // ---- prologue: stage tile 0 ----
#if HAVE_TDM
  if (wave == 0)
    tdm_load_2d((unsigned)(size_t)(void*)&Asl[0][0],
                &A[(long long)(bm*128)*lda], 32u, 128u,
                (unsigned long long)lda, /*pad every 16 DW*/3u);
#else
  #pragma unroll
  for (int rep = 0; rep < 8; ++rep){
    int e = (tid + rep*256)*2; int r = e >> 5, c = e & 31;
    areg[rep] = *(const unsigned int*)&A[(long long)(bm*128 + r)*lda + c];
  }
#endif
  #pragma unroll
  for (int rep = 0; rep < 8; ++rep){
    int e = (tid + rep*256)*2; int kk = e >> 7, nn = e & 127;
    breg[rep] = *(const unsigned int*)&B[(long long)kk*ldb + bn*128 + nn];
  }

  v8f acc[2][4] = {};
  for (int kt = 0; kt < nk; ++kt){
    const int buf = kt & 1;
    const int k0n = (kt + 1) << 5;
    // commit staged tile kt into LDS buffer `buf`
    #pragma unroll
    for (int rep = 0; rep < 8; ++rep){
      int e = (tid + rep*256)*2; int kk = e >> 7, nn = e & 127;
      Bsl[buf][nn*34 + kk]     = (unsigned short)(breg[rep] & 0xffffu);
      Bsl[buf][(nn+1)*34 + kk] = (unsigned short)(breg[rep] >> 16);
    }
#if !HAVE_TDM
    #pragma unroll
    for (int rep = 0; rep < 8; ++rep){
      int e = (tid + rep*256)*2; int r = e >> 5, c = e & 31;
      *(unsigned int*)&Asl[buf][r*34 + c] = areg[rep];
    }
#endif
#if HAVE_TDM
    if (wave == 0) __builtin_amdgcn_s_wait_tensorcnt(0);  // A tile kt landed
#endif
    __syncthreads();
    // kick off tile kt+1 while tile kt computes
    if (kt + 1 < nk){
#if HAVE_TDM
      if (wave == 0)
        tdm_load_2d((unsigned)(size_t)(void*)&Asl[buf^1][0],
                    &A[(long long)(bm*128)*lda + k0n], 32u, 128u,
                    (unsigned long long)lda, 3u);
#else
      #pragma unroll
      for (int rep = 0; rep < 8; ++rep){
        int e = (tid + rep*256)*2; int r = e >> 5, c = e & 31;
        areg[rep] = *(const unsigned int*)&A[(long long)(bm*128 + r)*lda + k0n + c];
      }
#endif
      #pragma unroll
      for (int rep = 0; rep < 8; ++rep){
        int e = (tid + rep*256)*2; int kk = e >> 7, nn = e & 127;
        breg[rep] = *(const unsigned int*)&B[(long long)(k0n + kk)*ldb + bn*128 + nn];
      }
      if (kt + 2 < nk && tid < 32)   // warm L2 for tile kt+2's B rows
        __builtin_prefetch(&B[(long long)((kt+2)*32 + tid)*ldb + bn*128], 0, 1);
    }
    // fragments + WMMA on buffer `buf`
    Frag16 af[2], bfr[4];
    #pragma unroll
    for (int sm = 0; sm < 2; ++sm){
      int m = wm*32 + sm*16 + l16;
      #pragma unroll
      for (int i = 0; i < 8; ++i)
        af[sm].u[i] = *(const unsigned int*)&Asl[buf][m*34 + pbA(i, half)];
    }
    #pragma unroll
    for (int sn = 0; sn < 4; ++sn){
      int n = wn*64 + sn*16 + l16;
      #pragma unroll
      for (int i = 0; i < 8; ++i)
        bfr[sn].u[i] = *(const unsigned int*)&Bsl[buf][n*34 + pbB(i, half)];
    }
    #pragma unroll
    for (int sm = 0; sm < 2; ++sm)
      #pragma unroll
      for (int sn = 0; sn < 4; ++sn)
        acc[sm][sn] = __builtin_amdgcn_wmma_f32_16x16x32_bf16(
            false, af[sm].v, false, bfr[sn].v, (short)0, acc[sm][sn], false, false);
  }
  // epilogue: uniform branches hoisted outside the store loops
  const long long cbase = (long long)bz*sC;
  if (Cf != nullptr){
    #pragma unroll
    for (int sm = 0; sm < 2; ++sm)
      #pragma unroll
      for (int sn = 0; sn < 4; ++sn)
        #pragma unroll
        for (int r = 0; r < 8; ++r){
          int row = bm*128 + wm*32 + sm*16 + r + 8*half;
          int col = bn*128 + wn*64 + sn*16 + l16;
          Cf[cbase + (long long)row*ldc + col] = acc[sm][sn][r];
        }
  }
  if (Cb != nullptr){
    #pragma unroll
    for (int sm = 0; sm < 2; ++sm)
      #pragma unroll
      for (int sn = 0; sn < 4; ++sn)
        #pragma unroll
        for (int r = 0; r < 8; ++r){
          int row = bm*128 + wm*32 + sm*16 + r + 8*half;
          int col = bn*128 + wn*64 + sn*16 + l16;
          Cb[cbase + (long long)row*ldc + col] = f2bf(acc[sm][sn][r]);
        }
  }
}

// ---------------- fused attention: scores->(mu,var)->r->ctx ----------------
__global__ __launch_bounds__(256) void attn_fused(
    const unsigned short* __restrict__ qh,       // [B][H][2048][64] bf16 (pre-scaled)
    const unsigned short* __restrict__ keys_hn,  // [B][H][512][64]
    const unsigned short* __restrict__ vals_t,   // [B][H][64][512]
    const float* __restrict__ w_mu, const float* __restrict__ w_sigma,
    unsigned short* __restrict__ ctx_bf)         // [B][2048][1024]
{
  extern __shared__ unsigned short smem[];
  unsigned short* keys_l = smem;                  // [512][66]
  unsigned short* vals_l = smem + 512*66;         // [64][514]
  unsigned short* rstage = vals_l + 64*514;       // [8 waves][16][34]

  const int tid = threadIdx.x;
  const int lane = tid & 31, wave = tid >> 5;
  const int half = lane >> 4, l16 = lane & 15;
  const int qb = blockIdx.x, h = blockIdx.y, b = blockIdx.z;
  const long long bh = (long long)(b*NH + h);
  const unsigned short* kg = keys_hn + bh * (NB*DH);
  const unsigned short* vg = vals_t  + bh * (DH*NB);

#if HAVE_TDM
  if (wave == 0){
    // keys: 512 rows x 64 bf16 (32 DW) -> padded stride 66; vals: 64 x 512 (256 DW) -> 514
    tdm_load_2d((unsigned)(size_t)(void*)keys_l, kg, 64u, 512u, 64ull, /*32 DW*/4u);
    tdm_load_2d((unsigned)(size_t)(void*)vals_l, vg, 512u, 64u, 512ull, /*256 DW*/7u);
    __builtin_amdgcn_s_wait_tensorcnt(0);
  }
#else
  for (int rep = 0; rep < 64; ++rep){            // keys -> LDS [n][66]
    int e = (tid + rep*256)*2;
    int n = e >> 6, c = e & 63;
    *(unsigned int*)&keys_l[n*66 + c] = *(const unsigned int*)&kg[e];
  }
  for (int rep = 0; rep < 64; ++rep){            // vals -> LDS [c][514]
    int e = (tid + rep*256)*2;
    int c = e >> 9, n = e & 511;
    *(unsigned int*)&vals_l[c*514 + n] = *(const unsigned int*)&vg[e];
  }
#endif
  __syncthreads();

  const int row0 = qb*128 + wave*16;
  const unsigned short* qg = qh + (bh * L_SEQ + row0) * DH;
  Frag16 qa[2];
  #pragma unroll
  for (int ks = 0; ks < 2; ++ks)
    #pragma unroll
    for (int i = 0; i < 8; ++i)
      qa[ks].u[i] = *(const unsigned int*)&qg[l16*DH + ks*32 + pbA(i, half)];

  float dmu[8], dsg[8];
  #pragma unroll
  for (int r = 0; r < 8; ++r){ dmu[r] = 0.f; dsg[r] = 0.f; }

  for (int j = 0; j < 32; ++j){                  // pass 1: scores + row dots
    int n = j*16 + l16;
    float wm = w_mu[n], wsv = w_sigma[n];
    v8f s = {};
    #pragma unroll
    for (int ks = 0; ks < 2; ++ks){
      Frag16 kb;
      #pragma unroll
      for (int i = 0; i < 8; ++i)
        kb.u[i] = *(const unsigned int*)&keys_l[n*66 + ks*32 + pbB(i, half)];
      s = __builtin_amdgcn_wmma_f32_16x16x32_bf16(false, qa[ks].v, false, kb.v, (short)0, s, false, false);
    }
    #pragma unroll
    for (int r = 0; r < 8; ++r){ dmu[r] += s[r]*wm; dsg[r] += s[r]*wsv; }
  }
  #pragma unroll
  for (int r = 0; r < 8; ++r){                   // reduce within 16-lane halves
    #pragma unroll
    for (int off = 1; off < 16; off <<= 1){
      dmu[r] += __shfl_xor(dmu[r], off, 32);
      dsg[r] += __shfl_xor(dsg[r], off, 32);
    }
  }
  float mu[8], var0[8];
  #pragma unroll
  for (int r = 0; r < 8; ++r){
    mu[r] = 1.0f / (1.0f + __expf(-dmu[r]));
    float x = dsg[r];
    float sp = (x > 20.f) ? x : log1pf(__expf(x));
    var0[r] = fmaxf(sp, 1e-6f);
  }

  unsigned short* rs = rstage + wave * (16*34);  // wave-private stage
  v8f cacc[4] = {};
  for (int chunk = 0; chunk < 16; ++chunk){      // pass 2: r elementwise + ctx GEMM
    int n0 = chunk*32;
    #pragma unroll
    for (int sub = 0; sub < 2; ++sub){
      int nn = n0 + sub*16 + l16;
      float mub = (float)(nn >> 1) * (1.0f/255.0f);
      float sgb = (nn & 1) ? 0.01f : 0.005f;
      float sg2 = sgb*sgb;
      #pragma unroll
      for (int r = 0; r < 8; ++r){
        float v = var0[r] + sg2;
        float d = mu[r] - mub;
        float rv = __expf(-0.5f*d*d/v) * (INV_SQRT_2PI * rsqrtf(v));
        rs[(r + 8*half)*34 + sub*16 + l16] = f2bf(rv);  // C-layout -> row-major stage
      }
    }
    Frag16 ra;                                   // re-read as A fragment (in-order DS)
    #pragma unroll
    for (int i = 0; i < 8; ++i)
      ra.u[i] = *(const unsigned int*)&rs[l16*34 + pbA(i, half)];
    #pragma unroll
    for (int tn = 0; tn < 4; ++tn){
      Frag16 vb;
      #pragma unroll
      for (int i = 0; i < 8; ++i)
        vb.u[i] = *(const unsigned int*)&vals_l[(tn*16 + l16)*514 + n0 + pbB(i, half)];
      cacc[tn] = __builtin_amdgcn_wmma_f32_16x16x32_bf16(false, ra.v, false, vb.v, (short)0, cacc[tn], false, false);
    }
  }
  #pragma unroll
  for (int tn = 0; tn < 4; ++tn){                // write ctx merged back to [B][L][D]
    int c = tn*16 + l16;
    #pragma unroll
    for (int r = 0; r < 8; ++r){
      int row = row0 + r + 8*half;
      ctx_bf[((long long)b*L_SEQ + row)*DMODEL + h*DH + c] = f2bf(cacc[tn][r]);
    }
  }
}

// ---------------- host launcher ----------------
extern "C" void kernel_launch(void* const* d_in, const int* in_sizes, int n_in,
                              void* d_out, int out_size, void* d_ws, size_t ws_size,
                              hipStream_t stream){
  (void)in_sizes; (void)n_in; (void)out_size; (void)ws_size;
  const float* k_in    = (const float*)d_in[0];
  const float* query   = (const float*)d_in[1];
  const float* Wq      = (const float*)d_in[2];
  const float* Wk      = (const float*)d_in[3];
  const float* Wv      = (const float*)d_in[4];
  const float* Wo      = (const float*)d_in[5];
  const float* w_mu    = (const float*)d_in[6];
  const float* w_sigma = (const float*)d_in[7];
  float* out = (float*)d_out;

  char* ws = (char*)d_ws;
  size_t o = 0;
  auto alloc = [&](size_t bytes)->char*{
    char* p = ws + o; o = (o + bytes + 255) & ~(size_t)255; return p;
  };
  const size_t BLD2 = (size_t)BATCH*L_SEQ*DMODEL*2;   // bf16 [B][L][D]
  const size_t BND2 = (size_t)BATCH*NB*DMODEL*2;      // bf16 [B][N][D]
  float* F            = (float*)alloc((size_t)NB*TWO_L*4);
  float* Aug          = (float*)alloc((size_t)NB*1024*4);
  unsigned short* GsT = (unsigned short*)alloc((size_t)NB*L_SEQ*2);
  unsigned short* kbf = (unsigned short*)alloc(BLD2);
  unsigned short* WkT = (unsigned short*)alloc((size_t)DMODEL*DMODEL*2);
  unsigned short* WvT = (unsigned short*)alloc((size_t)DMODEL*DMODEL*2);
  unsigned short* WqT = (unsigned short*)alloc((size_t)DMODEL*DMODEL*2);
  unsigned short* WoT = (unsigned short*)alloc((size_t)DMODEL*DMODEL*2);
  unsigned short* Bm      = (unsigned short*)alloc(BND2);
  unsigned short* keys    = (unsigned short*)alloc(BND2);
  unsigned short* vals    = (unsigned short*)alloc(BND2);
  unsigned short* keys_hn = (unsigned short*)alloc(BND2);
  unsigned short* vals_t  = (unsigned short*)alloc(BND2);
  unsigned short* qflat   = (unsigned short*)alloc(BLD2);
  unsigned short* qp      = (unsigned short*)alloc(BLD2);
  unsigned short* qh      = (unsigned short*)alloc(BLD2);
  unsigned short* ctx     = (unsigned short*)alloc(BLD2);

  const long long sLD = (long long)L_SEQ*DMODEL;
  const long long sND = (long long)NB*DMODEL;

  k_build_F<<<8192, 256, 0, stream>>>(F);
  k_gram   <<<2048, 256, 0, stream>>>(F, Aug);
  k_invert <<<1, 1024, 0, stream>>>(Aug);
  k_gsT    <<<4096, 256, 0, stream>>>(F, Aug, GsT);
  k_cvt    <<<65536, 256, 0, stream>>>(k_in, kbf, (long long)BATCH*L_SEQ*DMODEL);
  k_cvt_wT <<<4096, 256, 0, stream>>>(Wk, WkT, 1.0f);
  k_cvt_wT <<<4096, 256, 0, stream>>>(Wv, WvT, 1.0f);
  k_cvt_wT <<<4096, 256, 0, stream>>>(Wq, WqT, 0.125f);   // fold 1/sqrt(64)
  k_cvt_wT <<<4096, 256, 0, stream>>>(Wo, WoT, 1.0f);
  k_qflat  <<<65536, 256, 0, stream>>>(query, qflat);

  // Bmat[b] = GsT @ k[b] : [512,1024], K=2048
  gemm_bf16<<<dim3(8,4,8), 256, 0, stream>>>(GsT, kbf, nullptr, Bm,
      512, 1024, 2048, 2048, 1024, 1024, 0, sLD, sND);
  // keys/vals[b] = Bmat[b] @ W{k,v}^T : [512,1024], K=1024
  gemm_bf16<<<dim3(8,4,8), 256, 0, stream>>>(Bm, WkT, nullptr, keys,
      512, 1024, 1024, 1024, 1024, 1024, sND, 0, sND);
  gemm_bf16<<<dim3(8,4,8), 256, 0, stream>>>(Bm, WvT, nullptr, vals,
      512, 1024, 1024, 1024, 1024, 1024, sND, 0, sND);
  // q proj: qp[b] = qflat[b] @ (Wq^T/8) : [2048,1024], K=1024
  gemm_bf16<<<dim3(8,16,8), 256, 0, stream>>>(qflat, WqT, nullptr, qp,
      2048, 1024, 1024, 1024, 1024, 1024, sLD, 0, sLD);

  k_split <<<16384, 256, 0, stream>>>(keys, keys_hn, NB);
  k_splitT<<<16384, 256, 0, stream>>>(vals, vals_t);
  k_split <<<65536, 256, 0, stream>>>(qp, qh, L_SEQ);

  size_t shmem = (size_t)(512*66 + 64*514 + 8*16*34) * 2;  // ~139 KB
  attn_fused<<<dim3(16,16,8), 256, shmem, stream>>>(qh, keys_hn, vals_t, w_mu, w_sigma, ctx);

  // out[b] = ctx[b] @ Wo^T : f32 -> d_out
  gemm_bf16<<<dim3(8,16,8), 256, 0, stream>>>(ctx, WoT, out, nullptr,
      2048, 1024, 1024, 1024, 1024, 1024, sLD, 0, sLD);
}
